// LSTM_Build_74869869904701
// MI455X (gfx1250) — matrix operand order repaired and checked by reading, not compiled, not run on hardware
//
#include <hip/hip_runtime.h>
#include <hip/hip_bf16.h>

typedef __attribute__((ext_vector_type(16))) _Float16 v16h;
typedef __attribute__((ext_vector_type(4)))  _Float16 v4h;
typedef __attribute__((ext_vector_type(8)))  float    v8f;

#define B_TOT   4096
#define T_STEPS 80
#define E_DIM   100
#define E_PAD   128      // pad K to 4 * 32
#define U_DIM   64
#define G4      256      // 4*U gate width
#define M_TILE  16       // batch rows per workgroup
#define THREADS 512      // 16 waves (wave32)
#define XT_ACT  (M_TILE * (E_DIM / 4))   // 400 active stager threads

__device__ __forceinline__ float sigf(float x)     { return 1.0f / (1.0f + __expf(-x)); }
__device__ __forceinline__ float tanhfast(float x) { return 2.0f * sigf(2.0f * x) - 1.0f; }

extern "C" __global__ __launch_bounds__(THREADS)
void lstm_fused_cdna5(const int* __restrict__ tokens,
                      const float* __restrict__ emb,
                      const float* __restrict__ k1,
                      const float* __restrict__ r1,
                      const float* __restrict__ b1,
                      const float* __restrict__ Wd,
                      const float* __restrict__ bd,
                      float* __restrict__ out)
{
    extern __shared__ char smem[];
    _Float16* k1h  = (_Float16*)smem;                 // [E_PAD][G4]   64 KB
    _Float16* r1h  = k1h + E_PAD * G4;                // [U_DIM][G4]   32 KB
    _Float16* xth  = r1h + U_DIM * G4;                // [M_TILE][E_PAD] 4 KB
    _Float16* hh   = xth + M_TILE * E_PAD;            // [M_TILE][U_DIM] 2 KB
    float*    zz   = (float*)(hh + M_TILE * U_DIM);   // [M_TILE][G4]  16 KB
    float*    b1s  = zz + M_TILE * G4;                // [G4]           1 KB
    int*      toks = (int*)(b1s + G4);                // [M_TILE][T_STEPS] 5 KB

    const int tid   = threadIdx.x;
    const int lane  = tid & 31;
    const int wave  = tid >> 5;        // 0..15 -> N tile of the 256-wide gate matrix
    const int nbase = wave * 16;
    const int bbase = blockIdx.x * M_TILE;

    // ---- one-time preloads: weights (f16), bias, token block, zeroed h / xt-pad ----
    for (int idx = tid; idx < E_PAD * G4; idx += THREADS) {
        int k = idx / G4, n = idx - k * G4;
        k1h[idx] = (_Float16)(k < E_DIM ? k1[k * G4 + n] : 0.0f);
    }
    for (int idx = tid; idx < U_DIM * G4; idx += THREADS)
        r1h[idx] = (_Float16)r1[idx];
    for (int idx = tid; idx < G4; idx += THREADS)
        b1s[idx] = b1[idx];
    for (int idx = tid; idx < M_TILE * T_STEPS; idx += THREADS)
        toks[idx] = tokens[(bbase + idx / T_STEPS) * T_STEPS + (idx % T_STEPS)];
    for (int idx = tid; idx < M_TILE * U_DIM; idx += THREADS)
        hh[idx] = (_Float16)0.0f;
    for (int idx = tid; idx < M_TILE * E_PAD; idx += THREADS)
        xth[idx] = (_Float16)0.0f;     // cols 100..127 stay zero forever
    __syncthreads();

    // ---- hoist loop-invariant B fragments (ISA 16-bit B 32x16 layout) ----
    // lanes 0-15: N = lane, K = 0..7,16..23 ; lanes 16-31: N = lane-16, K = 8..15,24..31
    auto load_b = [&](const _Float16* w, int kbase) {
        v16h b;
        int nn   = nbase + (lane & 15);
        int koff = (lane < 16) ? 0 : 8;
        #pragma unroll
        for (int i = 0; i < 16; ++i)
            b[i] = w[(kbase + koff + (i < 8 ? i : i + 8)) * G4 + nn];
        return b;
    };
    const v16h bk0 = load_b(k1h, 0),  bk1 = load_b(k1h, 32);
    const v16h bk2 = load_b(k1h, 64), bk3 = load_b(k1h, 96);
    const v16h br0 = load_b(r1h, 0),  br1 = load_b(r1h, 32);

    // ISA 16-bit A 16x32 layout loader:
    // lane<16: halves 0..7 -> K+0..7, 8..15 -> K+16..23 ; lane>=16: same +8
    auto load_a = [&](const _Float16* src, int stride, int kbase) {
        v16h a;
        int m = lane & 15;
        const _Float16* p = src + m * stride + kbase + ((lane < 16) ? 0 : 8);
        #pragma unroll
        for (int i = 0; i < 8; ++i) a[i]     = p[i];
        #pragma unroll
        for (int i = 0; i < 8; ++i) a[8 + i] = p[16 + i];
        return a;
    };

    // per-thread cell state: 16x64 = 1024 (b,u) pairs over 512 threads -> 2 each
    float cA = 0.0f, cB = 0.0f;
    const int bA = tid >> 6,             uA = tid & 63;
    const int bB = (tid + THREADS) >> 6, uB = tid & 63;

    // ---- software-pipelined xt gather: float4 (16B aligned: row=400B, cg*4 mult of 16) ----
    const bool act  = tid < XT_ACT;
    const int  srow = act ? tid / (E_DIM / 4) : 0;       // 0..15
    const int  scg  = act ? (tid % (E_DIM / 4)) * 4 : 0; // col 0,4,...,96
    v4h xreg = {};
    if (act) {
        int tok = toks[srow * T_STEPS + 0];
        float4 f4 = *(const float4*)(emb + (long)tok * E_DIM + scg);
        xreg = v4h{(_Float16)f4.x, (_Float16)f4.y, (_Float16)f4.z, (_Float16)f4.w};
    }

    for (int t = 0; t < T_STEPS; ++t) {
        if (act)
            *(v4h*)(xth + srow * E_PAD + scg) = xreg;   // one ds_store_b64
        __syncthreads();

        // issue next step's gather now: HBM latency overlaps the WMMA chain below
        if (act && (t + 1 < T_STEPS)) {
            int tok = toks[srow * T_STEPS + t + 1];
            float4 f4 = *(const float4*)(emb + (long)tok * E_DIM + scg);
            xreg = v4h{(_Float16)f4.x, (_Float16)f4.y, (_Float16)f4.z, (_Float16)f4.w};
        }

        // ---- issue ALL A-fragment DS loads up front (12 x ds_load_b128),
        //      then run two independent WMMA accumulation chains ----
        v16h a0 = load_a(xth, E_PAD, 0);
        v16h a1 = load_a(xth, E_PAD, 32);
        v16h a2 = load_a(xth, E_PAD, 64);
        v16h a3 = load_a(xth, E_PAD, 96);
        v16h a4 = load_a(hh,  U_DIM, 0);
        v16h a5 = load_a(hh,  U_DIM, 32);

        v8f accX = {};   // xt @ k1 chain (4 WMMAs)
        v8f accH = {};   // h  @ r1 chain (2 WMMAs) — independent, overlaps accX chain
        accX = __builtin_amdgcn_wmma_f32_16x16x32_f16(false, a0, false, bk0, (short)0, accX, false, false);
        accH = __builtin_amdgcn_wmma_f32_16x16x32_f16(false, a4, false, br0, (short)0, accH, false, false);
        accX = __builtin_amdgcn_wmma_f32_16x16x32_f16(false, a1, false, bk1, (short)0, accX, false, false);
        accH = __builtin_amdgcn_wmma_f32_16x16x32_f16(false, a5, false, br1, (short)0, accH, false, false);
        accX = __builtin_amdgcn_wmma_f32_16x16x32_f16(false, a2, false, bk2, (short)0, accX, false, false);
        accX = __builtin_amdgcn_wmma_f32_16x16x32_f16(false, a3, false, bk3, (short)0, accX, false, false);

        // C/D layout: VGPR j -> M=j (lanes 0-15, N=lane) / M=j+8 (lanes 16-31, N=lane-16)
        {
            int n  = lane & 15;
            int mb = (lane < 16) ? 0 : 8;
            #pragma unroll
            for (int j = 0; j < 8; ++j)
                zz[(mb + j) * G4 + nbase + n] = accX[j] + accH[j];
        }
        __syncthreads();

        // ---- gates + state update (f32, fast transcendentals), h -> f16 LDS ----
        {
            float zi = zz[bA * G4 +       uA] + b1s[uA];
            float zf = zz[bA * G4 +  64 + uA] + b1s[64 + uA];
            float zg = zz[bA * G4 + 128 + uA] + b1s[128 + uA];
            float zo = zz[bA * G4 + 192 + uA] + b1s[192 + uA];
            cA = sigf(zf) * cA + sigf(zi) * tanhfast(zg);
            hh[bA * U_DIM + uA] = (_Float16)(sigf(zo) * tanhfast(cA));

            zi = zz[bB * G4 +       uB] + b1s[uB];
            zf = zz[bB * G4 +  64 + uB] + b1s[64 + uB];
            zg = zz[bB * G4 + 128 + uB] + b1s[128 + uB];
            zo = zz[bB * G4 + 192 + uB] + b1s[192 + uB];
            cB = sigf(zf) * cB + sigf(zi) * tanhfast(zg);
            hh[bB * U_DIM + uB] = (_Float16)(sigf(zo) * tanhfast(cB));
        }
        __syncthreads();
    }

    // ---- output: sigmoid(h1 @ Wd + bd) ----
    if (tid < M_TILE) {
        float accv = 0.0f;
        #pragma unroll 8
        for (int u = 0; u < U_DIM; ++u)
            accv += (float)hh[tid * U_DIM + u] * Wd[u];
        out[bbase + tid] = sigf(accv + bd[0]);
    }
}

extern "C" void kernel_launch(void* const* d_in, const int* in_sizes, int n_in,
                              void* d_out, int out_size, void* d_ws, size_t ws_size,
                              hipStream_t stream) {
    // setup_inputs order: tokens, emb, k0, r0, b0, k1, r1, b1, Wd, bd
    const int*   tokens = (const int*)  d_in[0];
    const float* emb    = (const float*)d_in[1];
    // d_in[2..4] = k0, r0, b0: dead — cell 0 never feeds the output
    const float* k1     = (const float*)d_in[5];
    const float* r1     = (const float*)d_in[6];
    const float* b1     = (const float*)d_in[7];
    const float* Wd     = (const float*)d_in[8];
    const float* bd     = (const float*)d_in[9];

    const size_t shmem =
        (size_t)E_PAD * G4 * sizeof(_Float16) +      // k1 f16
        (size_t)U_DIM * G4 * sizeof(_Float16) +      // r1 f16
        (size_t)M_TILE * E_PAD * sizeof(_Float16) +  // xt tile
        (size_t)M_TILE * U_DIM * sizeof(_Float16) +  // h state
        (size_t)M_TILE * G4 * sizeof(float) +        // z gates
        (size_t)G4 * sizeof(float) +                 // bias
        (size_t)M_TILE * T_STEPS * sizeof(int);      // token block

    lstm_fused_cdna5<<<B_TOT / M_TILE, THREADS, shmem, stream>>>(
        tokens, emb, k1, r1, b1, Wd, bd, (float*)d_out);
}
